// GINEEncoderEdgeUpd_60120952209608
// MI455X (gfx1250) — matrix operand order repaired and checked by reading, not compiled, not run on hardware
//
#include <hip/hip_runtime.h>

// ---------------- types ----------------
typedef __attribute__((ext_vector_type(16))) __bf16 bf16x16;
typedef __attribute__((ext_vector_type(8)))  __bf16 bf16x8;
typedef __attribute__((ext_vector_type(4)))  __bf16 bf16x4;
typedef __attribute__((ext_vector_type(8)))  float  v8f;

// ---------------- problem constants ----------------
#define NN_      10000
#define NE_      160000
#define NG_      64
#define NODE_IN_ 115
#define EDGE_IN_ 14
#define HD_      256
#define UHD_     64
#define DEPTH_   5
#define BN_EPS_  1e-5f
#define NODE_KP_ 128   // NODE_IN padded to 32
#define EDGE_KP_ 32    // EDGE_IN padded to 32

// ---------------- workspace layout (bytes) ----------------
static constexpr size_t OFF_XH   = 0;                                     // f32 [N,256]
static constexpr size_t OFF_AGG  = OFF_XH   + (size_t)NN_ * HD_ * 4;      // f32 [N,256]
static constexpr size_t OFF_XBF  = OFF_AGG  + (size_t)NN_ * HD_ * 4;      // bf16 [N,256]
static constexpr size_t OFF_EBF  = OFF_XBF  + (size_t)NN_ * HD_ * 2;      // bf16 [E,256]
static constexpr size_t OFF_POOL = OFF_EBF  + (size_t)NE_ * HD_ * 2;      // f32 [G,256]
static constexpr size_t OFF_CNT  = OFF_POOL + (size_t)NG_ * HD_ * 4;      // f32 [G]
static constexpr size_t OFF_XPAD = (OFF_CNT + (size_t)NG_ * 4 + 255) & ~(size_t)255;
static constexpr size_t OFF_EPAD = OFF_XPAD + (size_t)NN_ * NODE_KP_ * 4;
static constexpr size_t OFF_PW   = OFF_EPAD + (size_t)NE_ * EDGE_KP_ * 4;

// ---------------- WMMA helpers ----------------
__device__ __forceinline__ v8f wmma_bf16(bf16x16 a, bf16x16 b, v8f c) {
  return __builtin_amdgcn_wmma_f32_16x16x32_bf16(false, a, false, b, (short)0, c,
                                                 false, false);
}

// packed B fragment: 16 contiguous bf16 per lane (32B) -> two global_load_b128
__device__ __forceinline__ bf16x16 load_b_frag(const __bf16* __restrict__ pB,
                                               int frag, int lane) {
  return *(const bf16x16*)(pB + ((size_t)frag * 32 + lane) * 16);
}

__device__ __forceinline__ bf16x16 join8(bf16x8 l, bf16x8 h) {
  return __builtin_shufflevector(l, h, 0, 1, 2, 3, 4, 5, 6, 7,
                                 8, 9, 10, 11, 12, 13, 14, 15);
}

// A fragment straight from a bf16 row: two aligned 16B loads, no conversion.
// elems 0..7 = K0+half*8+j, elems 8..15 = K0+16+half*8+j (ISA 16-bit A layout)
__device__ __forceinline__ bf16x16 a_frag_bf16g(const __bf16* base, int half) {
  bf16x8 l = *(const bf16x8*)(base + half * 8);
  bf16x8 h = *(const bf16x8*)(base + 16 + half * 8);
  return join8(l, h);
}

// convert 4 raw float4 groups into one A fragment (for f32-resident A operands)
__device__ __forceinline__ bf16x16 cvt_frag(float4 a0, float4 a1, float4 b0, float4 b1) {
  bf16x16 r;
  r[0]=(__bf16)a0.x;  r[1]=(__bf16)a0.y;  r[2]=(__bf16)a0.z;  r[3]=(__bf16)a0.w;
  r[4]=(__bf16)a1.x;  r[5]=(__bf16)a1.y;  r[6]=(__bf16)a1.z;  r[7]=(__bf16)a1.w;
  r[8]=(__bf16)b0.x;  r[9]=(__bf16)b0.y;  r[10]=(__bf16)b0.z; r[11]=(__bf16)b0.w;
  r[12]=(__bf16)b1.x; r[13]=(__bf16)b1.y; r[14]=(__bf16)b1.z; r[15]=(__bf16)b1.w;
  return r;
}

__device__ __forceinline__ bf16x16 a_frag_f32(const float* base, int half) {
  const float4* q0 = (const float4*)(base + half * 8);
  const float4* q1 = (const float4*)(base + 16 + half * 8);
  return cvt_frag(q0[0], q0[1], q1[0], q1[1]);
}

// A fragment of (p + q) elementwise (for z = x + agg)
__device__ __forceinline__ bf16x16 a_frag_sum(const float* p, const float* q, int half) {
  const float4* pa = (const float4*)(p + half * 8);
  const float4* pb = (const float4*)(p + 16 + half * 8);
  const float4* qa = (const float4*)(q + half * 8);
  const float4* qb = (const float4*)(q + 16 + half * 8);
  float4 a0 = pa[0], a1 = pa[1], b0 = pb[0], b1 = pb[1];
  float4 c0 = qa[0], c1 = qa[1], d0 = qb[0], d1 = qb[1];
  a0.x += c0.x; a0.y += c0.y; a0.z += c0.z; a0.w += c0.w;
  a1.x += c1.x; a1.y += c1.y; a1.z += c1.z; a1.w += c1.w;
  b0.x += d0.x; b0.y += d0.y; b0.z += d0.z; b0.w += d0.w;
  b1.x += d1.x; b1.y += d1.y; b1.z += d1.z; b1.w += d1.w;
  return cvt_frag(a0, a1, b0, b1);
}

// A fragment from bf16 staged in LDS (two aligned 16B groups -> ds_load_b128)
__device__ __forceinline__ bf16x16 a_frag_lds(const __bf16* p0, const __bf16* p1) {
  bf16x8 l = *(const bf16x8*)p0;
  bf16x8 h = *(const bf16x8*)p1;
  return join8(l, h);
}

// ---------------- A padding: f32 [rows x Kin] -> f32 [rows x Kpad] zero-padded --
__global__ void pad_a_kernel(const float* __restrict__ src, float* __restrict__ dst,
                             int rows, int Kin, int Kpad) {
  size_t idx = (size_t)blockIdx.x * blockDim.x + threadIdx.x;
  if (idx >= (size_t)rows * Kpad) return;
  int r = (int)(idx / Kpad);
  int k = (int)(idx % Kpad);
  dst[idx] = (k < Kin) ? src[(size_t)r * Kin + k] : 0.0f;
}

// ---------------- weight packing: row-major f32 [K x Nw] -> bf16 B fragments ----
__global__ void pack_b_kernel(const float* __restrict__ w, __bf16* __restrict__ out,
                              int K, int Nw, int KB, int NT) {
  int idx = blockIdx.x * blockDim.x + threadIdx.x;
  int total = KB * NT * 512;
  if (idx >= total) return;
  int j = idx & 15;
  int lane = (idx >> 4) & 31;
  int rem = idx >> 9;
  int nt = rem % NT;
  int kb = rem / NT;
  int k = kb * 32 + (j >> 3) * 16 + (lane >> 4) * 8 + (j & 7);
  int n = nt * 16 + (lane & 15);
  float v = (k < K) ? w[(size_t)k * Nw + n] : 0.0f;
  out[idx] = (__bf16)v;
}

// ------- generic GEMM: padded f32 A [rows x (KB*32)] x packedB -> [rows x 256]
// OUTF32: write f32 C; OUTBF: write bf16 Cbf mirror
template <int KB, int RELU, int OUTF32, int OUTBF>
__global__ void gemm_bias256_kernel(const float* __restrict__ A,
                                    const float* __restrict__ bias,
                                    const __bf16* __restrict__ pB,
                                    float* __restrict__ C,
                                    __bf16* __restrict__ Cbf, int rows) {
  int wid = (blockIdx.x * blockDim.x + threadIdx.x) >> 5;
  int lane = threadIdx.x & 31;
  int tiles = rows >> 4;
  if (wid >= tiles * 16) return;
  int rt = wid >> 4, nt = wid & 15;
  int half = lane >> 4, mn = lane & 15;
  const float* row = A + (size_t)(rt * 16 + mn) * (KB * 32);
  v8f acc = {0.f, 0.f, 0.f, 0.f, 0.f, 0.f, 0.f, 0.f};
#pragma unroll
  for (int kb = 0; kb < KB; ++kb) {
    bf16x16 a = a_frag_f32(row + kb * 32, half);
    bf16x16 b = load_b_frag(pB, kb * 16 + nt, lane);
    acc = wmma_bf16(a, b, acc);
  }
#pragma unroll
  for (int r = 0; r < 8; ++r) {
    int ro = rt * 16 + r + 8 * half;
    int co = nt * 16 + mn;
    float v = acc[r] + bias[co];
    if (RELU) v = fmaxf(v, 0.f);
    size_t o = (size_t)ro * HD_ + co;
    if (OUTF32) C[o] = v;
    if (OUTBF)  Cbf[o] = (__bf16)v;
  }
}

// ---------------- EdgeUpdate: e += MLP([x[src], x[dst], e]) -----------------
// one wave per 16 edges; all A operands load directly as bf16 (2x b128, no cvt)
__global__ void edge_update_kernel(const __bf16* __restrict__ xbf, __bf16* ebf,
                                   const int* __restrict__ src,
                                   const int* __restrict__ dst,
                                   const __bf16* __restrict__ pw1,
                                   const float* __restrict__ b1,
                                   const __bf16* __restrict__ pw2,
                                   const float* __restrict__ b2) {
  extern __shared__ char smem[];
  int wib = threadIdx.x >> 5;
  int lane = threadIdx.x & 31;
  int t = blockIdx.x * 8 + wib;
  if (t >= NE_ / 16) return;
  int half = lane >> 4, mn = lane & 15;
  int eid = t * 16 + mn;
  int s = src[eid], d = dst[eid];
  const __bf16* xs = xbf + (size_t)s * HD_;
  const __bf16* xd = xbf + (size_t)d * HD_;
  const __bf16* er = ebf + (size_t)eid * HD_;
  __bf16* myh = (__bf16*)smem + wib * (16 * UHD_);

  v8f acc0 = {0.f,0.f,0.f,0.f,0.f,0.f,0.f,0.f};
  v8f acc1 = acc0, acc2 = acc0, acc3 = acc0;
#pragma unroll
  for (int seg = 0; seg < 3; ++seg) {
    const __bf16* segp = (seg == 0) ? xs : (seg == 1) ? xd : er;
#pragma unroll
    for (int kk = 0; kk < 8; ++kk) {
      int kb = seg * 8 + kk;
      bf16x16 a = a_frag_bf16g(segp + kk * 32, half);
      acc0 = wmma_bf16(a, load_b_frag(pw1, kb * 4 + 0, lane), acc0);
      acc1 = wmma_bf16(a, load_b_frag(pw1, kb * 4 + 1, lane), acc1);
      acc2 = wmma_bf16(a, load_b_frag(pw1, kb * 4 + 2, lane), acc2);
      acc3 = wmma_bf16(a, load_b_frag(pw1, kb * 4 + 3, lane), acc3);
    }
  }
  // h = relu(acc + b1) -> LDS bf16 [16][64]
#pragma unroll
  for (int r = 0; r < 8; ++r) {
    int mr = r + 8 * half;
    myh[mr * UHD_ +  0 + mn] = (__bf16)fmaxf(acc0[r] + b1[ 0 + mn], 0.f);
    myh[mr * UHD_ + 16 + mn] = (__bf16)fmaxf(acc1[r] + b1[16 + mn], 0.f);
    myh[mr * UHD_ + 32 + mn] = (__bf16)fmaxf(acc2[r] + b1[32 + mn], 0.f);
    myh[mr * UHD_ + 48 + mn] = (__bf16)fmaxf(acc3[r] + b1[48 + mn], 0.f);
  }
  // GEMM2: A frags from LDS (K=64 -> 2 kb)
  bf16x16 a0 = a_frag_lds(myh + mn * UHD_ + half * 8,
                          myh + mn * UHD_ + 16 + half * 8);
  bf16x16 a1 = a_frag_lds(myh + mn * UHD_ + 32 + half * 8,
                          myh + mn * UHD_ + 48 + half * 8);
  for (int nt = 0; nt < 16; ++nt) {
    v8f acc = {0.f,0.f,0.f,0.f,0.f,0.f,0.f,0.f};
    acc = wmma_bf16(a0, load_b_frag(pw2, 0 * 16 + nt, lane), acc);
    acc = wmma_bf16(a1, load_b_frag(pw2, 1 * 16 + nt, lane), acc);
#pragma unroll
    for (int r = 0; r < 8; ++r) {
      int mr = r + 8 * half;
      int co = nt * 16 + mn;
      size_t o = (size_t)(t * 16 + mr) * HD_ + co;
      float ev = (float)ebf[o] + acc[r] + b2[co];
      ebf[o] = (__bf16)ev;
    }
  }
}

// ---------------- message + segment-sum scatter (bf16 reads, f32 atomics) -----
__global__ void msg_agg_kernel(const __bf16* __restrict__ xbf,
                               const __bf16* __restrict__ ebf,
                               const int* __restrict__ src,
                               const int* __restrict__ dst,
                               float* __restrict__ agg) {
  size_t idx = (size_t)blockIdx.x * blockDim.x + threadIdx.x;
  if (idx >= (size_t)NE_ * 64) return;
  int eid = (int)(idx >> 6);
  int c4 = ((int)idx & 63) << 2;
  int s = src[eid], d = dst[eid];
  bf16x4 xv = *(const bf16x4*)(xbf + (size_t)s * HD_ + c4);
  bf16x4 ev = *(const bf16x4*)(ebf + (size_t)eid * HD_ + c4);
  float m0 = fmaxf((float)xv[0] + (float)ev[0], 0.f);
  float m1 = fmaxf((float)xv[1] + (float)ev[1], 0.f);
  float m2 = fmaxf((float)xv[2] + (float)ev[2], 0.f);
  float m3 = fmaxf((float)xv[3] + (float)ev[3], 0.f);
  float* ag = agg + (size_t)d * HD_ + c4;
  __hip_atomic_fetch_add(&ag[0], m0, __ATOMIC_RELAXED, __HIP_MEMORY_SCOPE_AGENT);
  __hip_atomic_fetch_add(&ag[1], m1, __ATOMIC_RELAXED, __HIP_MEMORY_SCOPE_AGENT);
  __hip_atomic_fetch_add(&ag[2], m2, __ATOMIC_RELAXED, __HIP_MEMORY_SCOPE_AGENT);
  __hip_atomic_fetch_add(&ag[3], m3, __ATOMIC_RELAXED, __HIP_MEMORY_SCOPE_AGENT);
}

// ---------------- GINEConv MLP + BN + residual (in-place on x, + bf16 mirror) --
__global__ void conv_mlp_kernel(float* __restrict__ xh, __bf16* __restrict__ xbf,
                                const float* __restrict__ agg,
                                const __bf16* __restrict__ pw1,
                                const float* __restrict__ b1,
                                const __bf16* __restrict__ pw2,
                                const float* __restrict__ b2,
                                const float* __restrict__ bg,
                                const float* __restrict__ bb,
                                const float* __restrict__ bm,
                                const float* __restrict__ bv) {
  extern __shared__ char smem[];
  int wib = threadIdx.x >> 5;
  int lane = threadIdx.x & 31;
  int t = blockIdx.x * 8 + wib;
  if (t >= NN_ / 16) return;
  int half = lane >> 4, mn = lane & 15;
  const float* xr = xh + (size_t)(t * 16 + mn) * HD_;
  const float* ar = agg + (size_t)(t * 16 + mn) * HD_;
  __bf16* myh = (__bf16*)smem + wib * (16 * HD_);

  bf16x16 af[8];
#pragma unroll
  for (int kb = 0; kb < 8; ++kb) af[kb] = a_frag_sum(xr + kb * 32, ar + kb * 32, half);

  // GEMM1: h = relu(z@w1 + b1) -> LDS bf16 [16][256]
  for (int nt = 0; nt < 16; ++nt) {
    v8f acc = {0.f,0.f,0.f,0.f,0.f,0.f,0.f,0.f};
#pragma unroll
    for (int kb = 0; kb < 8; ++kb)
      acc = wmma_bf16(af[kb], load_b_frag(pw1, kb * 16 + nt, lane), acc);
#pragma unroll
    for (int r = 0; r < 8; ++r) {
      int mr = r + 8 * half;
      int co = nt * 16 + mn;
      myh[mr * HD_ + co] = (__bf16)fmaxf(acc[r] + b1[co], 0.f);
    }
  }
  // GEMM2 A frags from LDS
#pragma unroll
  for (int kb = 0; kb < 8; ++kb)
    af[kb] = a_frag_lds(myh + mn * HD_ + kb * 32 + half * 8,
                        myh + mn * HD_ + kb * 32 + 16 + half * 8);
  // GEMM2 + BN + residual relu; update f32 master and bf16 mirror
  for (int nt = 0; nt < 16; ++nt) {
    v8f acc = {0.f,0.f,0.f,0.f,0.f,0.f,0.f,0.f};
#pragma unroll
    for (int kb = 0; kb < 8; ++kb)
      acc = wmma_bf16(af[kb], load_b_frag(pw2, kb * 16 + nt, lane), acc);
#pragma unroll
    for (int r = 0; r < 8; ++r) {
      int mr = r + 8 * half;
      int co = nt * 16 + mn;
      float z = acc[r] + b2[co];
      z = (z - bm[co]) * rsqrtf(bv[co] + BN_EPS_) * bg[co] + bb[co];
      size_t o = (size_t)(t * 16 + mr) * HD_ + co;
      float nv = xh[o] + fmaxf(z, 0.f);
      xh[o] = nv;
      xbf[o] = (__bf16)nv;
    }
  }
}

// ---------------- global mean pool ----------------
__global__ void pool_sum_kernel(const float* __restrict__ xh,
                                const int* __restrict__ batch,
                                float* __restrict__ pooled, float* __restrict__ cnt) {
  size_t idx = (size_t)blockIdx.x * blockDim.x + threadIdx.x;
  if (idx >= (size_t)NN_ * 64) return;
  int n = (int)(idx >> 6);
  int c4 = ((int)idx & 63) << 2;
  int g = batch[n];
  float4 v = *(const float4*)(xh + (size_t)n * HD_ + c4);
  float* p = pooled + (size_t)g * HD_ + c4;
  __hip_atomic_fetch_add(&p[0], v.x, __ATOMIC_RELAXED, __HIP_MEMORY_SCOPE_AGENT);
  __hip_atomic_fetch_add(&p[1], v.y, __ATOMIC_RELAXED, __HIP_MEMORY_SCOPE_AGENT);
  __hip_atomic_fetch_add(&p[2], v.z, __ATOMIC_RELAXED, __HIP_MEMORY_SCOPE_AGENT);
  __hip_atomic_fetch_add(&p[3], v.w, __ATOMIC_RELAXED, __HIP_MEMORY_SCOPE_AGENT);
  if (c4 == 0)
    __hip_atomic_fetch_add(&cnt[g], 1.0f, __ATOMIC_RELAXED, __HIP_MEMORY_SCOPE_AGENT);
}

__global__ void pool_div_kernel(float* __restrict__ pooled,
                                const float* __restrict__ cnt) {
  int idx = blockIdx.x * blockDim.x + threadIdx.x;
  if (idx >= NG_ * HD_) return;
  int g = idx >> 8;
  pooled[idx] = pooled[idx] / fmaxf(cnt[g], 1.0f);
}

// ---------------- host launch ----------------
extern "C" void kernel_launch(void* const* d_in, const int* in_sizes, int n_in,
                              void* d_out, int out_size, void* d_ws, size_t ws_size,
                              hipStream_t stream) {
  (void)in_sizes; (void)n_in; (void)out_size; (void)ws_size;
  const float* x     = (const float*)d_in[0];
  const int*   eix   = (const int*)d_in[1];
  const float* ea    = (const float*)d_in[2];
  const int*   batch = (const int*)d_in[3];
  const float* xp_w  = (const float*)d_in[4];
  const float* xp_b  = (const float*)d_in[5];
  const float* ep_w  = (const float*)d_in[6];
  const float* ep_b  = (const float*)d_in[7];
  const float* u_w1  = (const float*)d_in[8];
  const float* u_b1  = (const float*)d_in[9];
  const float* u_w2  = (const float*)d_in[10];
  const float* u_b2  = (const float*)d_in[11];
  const float* c_w1  = (const float*)d_in[12];
  const float* c_b1  = (const float*)d_in[13];
  const float* c_w2  = (const float*)d_in[14];
  const float* c_b2  = (const float*)d_in[15];
  const float* bn_g  = (const float*)d_in[16];
  const float* bn_b  = (const float*)d_in[17];
  const float* bn_m  = (const float*)d_in[18];
  const float* bn_v  = (const float*)d_in[19];
  const float* ro_w  = (const float*)d_in[20];
  const float* ro_b  = (const float*)d_in[21];
  const int* srcI = eix;
  const int* dstI = eix + NE_;

  char* ws = (char*)d_ws;
  float*  xh     = (float*)(ws + OFF_XH);
  float*  agg    = (float*)(ws + OFF_AGG);
  __bf16* xbf    = (__bf16*)(ws + OFF_XBF);
  __bf16* ebf    = (__bf16*)(ws + OFF_EBF);
  float*  pooled = (float*)(ws + OFF_POOL);
  float*  cnt    = (float*)(ws + OFF_CNT);
  float*  xpad   = (float*)(ws + OFF_XPAD);
  float*  epad   = (float*)(ws + OFF_EPAD);
  __bf16* pw     = (__bf16*)(ws + OFF_PW);
  __bf16* pw_xp = pw;                               // 4*16*512
  __bf16* pw_ep = pw_xp + (size_t)4 * 16 * 512;     // 1*16*512
  __bf16* pw_ro = pw_ep + (size_t)1 * 16 * 512;     // 8*16*512
  __bf16* pw_u1 = pw_ro + (size_t)8 * 16 * 512;     // 5 * 24*4*512
  __bf16* pw_u2 = pw_u1 + (size_t)DEPTH_ * 24 * 4 * 512;
  __bf16* pw_c1 = pw_u2 + (size_t)DEPTH_ * 2 * 16 * 512;
  __bf16* pw_c2 = pw_c1 + (size_t)DEPTH_ * 8 * 16 * 512;

  auto pack = [&](const float* w, __bf16* o, int K, int Nw, int KB) {
    int NT = Nw / 16;
    int total = KB * NT * 512;
    pack_b_kernel<<<dim3((total + 255) / 256), dim3(256), 0, stream>>>(w, o, K, Nw,
                                                                       KB, NT);
  };
  pack(xp_w, pw_xp, NODE_IN_, HD_, 4);
  pack(ep_w, pw_ep, EDGE_IN_, HD_, 1);
  pack(ro_w, pw_ro, HD_, HD_, 8);
  for (int dep = 0; dep < DEPTH_; ++dep) {
    pack(u_w1 + (size_t)dep * 3 * HD_ * UHD_, pw_u1 + (size_t)dep * 24 * 4 * 512,
         3 * HD_, UHD_, 24);
    pack(u_w2 + (size_t)dep * UHD_ * HD_, pw_u2 + (size_t)dep * 2 * 16 * 512,
         UHD_, HD_, 2);
    pack(c_w1 + (size_t)dep * HD_ * HD_, pw_c1 + (size_t)dep * 8 * 16 * 512,
         HD_, HD_, 8);
    pack(c_w2 + (size_t)dep * HD_ * HD_, pw_c2 + (size_t)dep * 8 * 16 * 512,
         HD_, HD_, 8);
  }

  // zero-pad ragged A matrices so projection GEMMs use branch-free aligned loads
  {
    size_t tot = (size_t)NN_ * NODE_KP_;
    pad_a_kernel<<<dim3((unsigned)((tot + 255) / 256)), dim3(256), 0, stream>>>(
        x, xpad, NN_, NODE_IN_, NODE_KP_);
    tot = (size_t)NE_ * EDGE_KP_;
    pad_a_kernel<<<dim3((unsigned)((tot + 255) / 256)), dim3(256), 0, stream>>>(
        ea, epad, NE_, EDGE_IN_, EDGE_KP_);
  }

  // projections: node -> f32 + bf16 mirror; edge -> bf16 only
  {
    int waves = (NN_ / 16) * 16;
    gemm_bias256_kernel<4, 0, 1, 1>
        <<<dim3((waves * 32 + 255) / 256), dim3(256), 0, stream>>>(xpad, xp_b, pw_xp,
                                                                   xh, xbf, NN_);
    waves = (NE_ / 16) * 16;
    gemm_bias256_kernel<1, 0, 0, 1>
        <<<dim3((waves * 32 + 255) / 256), dim3(256), 0, stream>>>(epad, ep_b, pw_ep,
                                                                   (float*)nullptr,
                                                                   ebf, NE_);
  }

  for (int dep = 0; dep < DEPTH_; ++dep) {
    edge_update_kernel<<<dim3(NE_ / 16 / 8), dim3(256), 8 * 16 * UHD_ * 2, stream>>>(
        xbf, ebf, srcI, dstI,
        pw_u1 + (size_t)dep * 24 * 4 * 512, u_b1 + dep * UHD_,
        pw_u2 + (size_t)dep * 2 * 16 * 512, u_b2 + dep * HD_);
    hipMemsetAsync(agg, 0, (size_t)NN_ * HD_ * 4, stream);
    msg_agg_kernel<<<dim3((NE_ * 64) / 256), dim3(256), 0, stream>>>(xbf, ebf, srcI,
                                                                     dstI, agg);
    conv_mlp_kernel<<<dim3((NN_ / 16 + 7) / 8), dim3(256), 8 * 16 * HD_ * 2, stream>>>(
        xh, xbf, agg,
        pw_c1 + (size_t)dep * 8 * 16 * 512, c_b1 + dep * HD_,
        pw_c2 + (size_t)dep * 8 * 16 * 512, c_b2 + dep * HD_,
        bn_g + dep * HD_, bn_b + dep * HD_, bn_m + dep * HD_, bn_v + dep * HD_);
  }

  hipMemsetAsync(pooled, 0, (size_t)(NG_ * HD_ + NG_) * 4, stream);
  pool_sum_kernel<<<dim3((NN_ * 64 + 255) / 256), dim3(256), 0, stream>>>(xh, batch,
                                                                          pooled, cnt);
  pool_div_kernel<<<dim3((NG_ * HD_ + 255) / 256), dim3(256), 0, stream>>>(pooled, cnt);
  {
    int waves = (NG_ / 16) * 16;
    gemm_bias256_kernel<8, 1, 1, 0>
        <<<dim3((waves * 32 + 255) / 256), dim3(256), 0, stream>>>(
            pooled, ro_b, pw_ro, (float*)d_out, (__bf16*)nullptr, NG_);
  }
}
